// EmbeddingMatchLayer_80779744903535
// MI455X (gfx1250) — matrix operand order repaired and checked by reading, not compiled, not run on hardware
//
#include <hip/hip_runtime.h>
#include <hip/hip_bf16.h>
#include <math.h>

typedef __attribute__((ext_vector_type(16))) __bf16 v16bf;
typedef __attribute__((ext_vector_type(8)))  float  v8f;

#define R_      2048      // B*P
#define Q_      50000
#define QTILES  3125      // Q/16
#define QPAIRS  1562      // floor(QTILES/2); tile 3124 is the remainder
#define DK      64
#define TOPK    8

// workspace layout (bytes, all offsets 256-aligned)
#define OFF_SPROJ   0u          // f32  [Q][64]              12,800,000 B
#define OFF_SNORM   12800000u   // bf16 packed B-operand      6,400,000 B
#define OFF_TNORM   19200000u   // bf16 packed A-operand        262,144 B
#define OFF_TOPIDX  19462144u   // i32  [R][8]                   65,536 B

__device__ __forceinline__ unsigned short f2bf(float f) {
    unsigned u = __builtin_bit_cast(unsigned, f);
    u += 0x7fffu + ((u >> 16) & 1u);          // round-to-nearest-even
    return (unsigned short)(u >> 16);
}

// ---------------------------------------------------------------------------
// Kernel 1: s = src @ Ws^T + bs ; store f32 row and bf16 normalized row packed
// into WMMA B-operand (32x16, K x N) lanes in this code's contiguous k order (same as A):
//   lane = (Klocal/16)*16 + n ,  element j = Klocal%16 ,  kfrag = k/32
// ---------------------------------------------------------------------------
__global__ __launch_bounds__(64) void proj_source(const float* __restrict__ src,
                                                  const float* __restrict__ Ws,
                                                  const float* __restrict__ bs,
                                                  float* __restrict__ sProj,
                                                  unsigned short* __restrict__ sNorm) {
    const int q = blockIdx.x;
    const int k = threadIdx.x;                 // 0..63 output feature
    const float4* s4 = (const float4*)(src + (size_t)q * 768);
    const float4* w4 = (const float4*)(Ws  + (size_t)k * 768);
    float acc = 0.f;
#pragma unroll 4
    for (int d = 0; d < 192; ++d) {
        float4 a = s4[d], b = w4[d];
        acc = fmaf(a.x, b.x, acc); acc = fmaf(a.y, b.y, acc);
        acc = fmaf(a.z, b.z, acc); acc = fmaf(a.w, b.w, acc);
    }
    acc += bs[k];
    sProj[(size_t)q * 64 + k] = acc;

    __shared__ float red[64];
    red[k] = acc * acc;
    __syncthreads();
    for (int s = 32; s > 0; s >>= 1) { if (k < s) red[k] += red[k + s]; __syncthreads(); }
    const float rn = 1.0f / fmaxf(sqrtf(red[0]), 1e-12f);

    const int tile = q >> 4, n = q & 15;
    const int kf = k >> 5, kl = k & 31;
    const int half = kl >> 4, j = kl & 15;
    const int lane = half * 16 + n;
    sNorm[(size_t)tile * 1024 + kf * 512 + lane * 16 + j] = f2bf(acc * rn);
}

// ---------------------------------------------------------------------------
// Kernel 2: t = tgt @ Wt^T + bt ; normalized bf16 packed into WMMA A-operand
// (16x32, M x K) lanes in this code's contiguous k order (same as B): lane = half*16 + m,
//   Klocal = half*16 + j  ->  half=Kl>>4, j=Kl&15
// ---------------------------------------------------------------------------
__global__ __launch_bounds__(64) void proj_target(const float* __restrict__ tgt,
                                                  const float* __restrict__ Wt,
                                                  const float* __restrict__ bt,
                                                  unsigned short* __restrict__ tNorm) {
    const int r = blockIdx.x;                  // 0..2047 target row
    const int k = threadIdx.x;                 // 0..63
    const float4* s4 = (const float4*)(tgt + (size_t)r * 512);
    const float4* w4 = (const float4*)(Wt  + (size_t)k * 512);
    float acc = 0.f;
#pragma unroll 4
    for (int d = 0; d < 128; ++d) {
        float4 a = s4[d], b = w4[d];
        acc = fmaf(a.x, b.x, acc); acc = fmaf(a.y, b.y, acc);
        acc = fmaf(a.z, b.z, acc); acc = fmaf(a.w, b.w, acc);
    }
    acc += bt[k];

    __shared__ float red[64];
    red[k] = acc * acc;
    __syncthreads();
    for (int s = 32; s > 0; s >>= 1) { if (k < s) red[k] += red[k + s]; __syncthreads(); }
    const float rn = 1.0f / fmaxf(sqrtf(red[0]), 1e-12f);

    const int tile = r >> 4, m = r & 15;
    const int kf = k >> 5, kl = k & 31;
    const int half = kl >> 4;
    const int j = kl & 15;
    const int lane = half * 16 + m;
    tNorm[(size_t)tile * 1024 + kf * 512 + lane * 16 + j] = f2bf(acc * rn);
}

// ---------------------------------------------------------------------------
// Kernel 3: fused sim GEMM (bf16 WMMA, K=64 -> 2 x 16x16x32) + exact top-8.
// 1 block = 16 target rows, 8 waves split Q; two Q-tiles per iteration
// (4 WMMAs, 8 x b128 loads) to amortize loop overhead. Each lane keeps a
// sorted top-8 per C-row; merged through LDS in two 64KB phases.
// ---------------------------------------------------------------------------
__global__ __launch_bounds__(256) void sim_topk(const unsigned short* __restrict__ tNorm,
                                                const unsigned short* __restrict__ sNorm,
                                                int* __restrict__ topIdx) {
    __shared__ float cv[16 * 4 * 16 * 8];      // 32 KB
    __shared__ int   ci[16 * 4 * 16 * 8];      // 32 KB
    const int rowtile = blockIdx.x;
    const int wave = threadIdx.x >> 5;
    const int lane = threadIdx.x & 31;

    const v16bf* aB = (const v16bf*)(tNorm + (size_t)rowtile * 1024);
    const v16bf a0 = aB[lane];                 // kfrag 0 (K=0..31)
    const v16bf a1 = aB[32 + lane];            // kfrag 1 (K=32..63)

    float bv[8][8]; int bi[8][8];
#pragma unroll
    for (int r = 0; r < 8; ++r)
#pragma unroll
        for (int j = 0; j < 8; ++j) { bv[r][j] = -__builtin_inff(); bi[r][j] = 0; }

    // sorted insert of one 16x16 C tile into the per-lane top-8 lists
    auto insertC = [&](const v8f& c, int qi) {
#pragma unroll
        for (int r = 0; r < 8; ++r) {
            const float v = c[r];
            if (v > bv[r][7]) {                // rare after warm-up -> EXEC-skipped
                bv[r][7] = v; bi[r][7] = qi;
#pragma unroll
                for (int j = 7; j > 0; --j) {
                    if (bv[r][j] > bv[r][j - 1]) {
                        float t0 = bv[r][j]; bv[r][j] = bv[r][j - 1]; bv[r][j - 1] = t0;
                        int   t1 = bi[r][j]; bi[r][j] = bi[r][j - 1]; bi[r][j - 1] = t1;
                    }
                }
            }
        }
    };

    for (int p = wave; p < QPAIRS; p += 8) {   // tiles 2p and 2p+1 (contig. 4KB)
        const v16bf* bB = (const v16bf*)(sNorm + (size_t)p * 2048);
        const v16bf b0x = bB[lane];
        const v16bf b1x = bB[32 + lane];
        const v16bf b0y = bB[64 + lane];
        const v16bf b1y = bB[96 + lane];
        // unconditional speculative prefetch of the next pair into near caches
        __builtin_prefetch(sNorm + (size_t)(p + 8) * 2048 + lane * 64, 0, 3);

        v8f cx = {0.f, 0.f, 0.f, 0.f, 0.f, 0.f, 0.f, 0.f};
        cx = __builtin_amdgcn_wmma_f32_16x16x32_bf16(false, a0, false, b0x, (short)0, cx, false, false);
        cx = __builtin_amdgcn_wmma_f32_16x16x32_bf16(false, a1, false, b1x, (short)0, cx, false, false);
        v8f cy = {0.f, 0.f, 0.f, 0.f, 0.f, 0.f, 0.f, 0.f};
        cy = __builtin_amdgcn_wmma_f32_16x16x32_bf16(false, a0, false, b0y, (short)0, cy, false, false);
        cy = __builtin_amdgcn_wmma_f32_16x16x32_bf16(false, a1, false, b1y, (short)0, cy, false, false);

        const int qi = p * 32 + (lane & 15);
        insertC(cx, qi);
        insertC(cy, qi + 16);
    }
    if (wave == 0) {                           // leftover tile (QTILES is odd)
        const int tile = QTILES - 1;
        const v16bf* bB = (const v16bf*)(sNorm + (size_t)tile * 1024);
        const v16bf b0 = bB[lane];
        const v16bf b1 = bB[32 + lane];
        v8f c = {0.f, 0.f, 0.f, 0.f, 0.f, 0.f, 0.f, 0.f};
        c = __builtin_amdgcn_wmma_f32_16x16x32_bf16(false, a0, false, b0, (short)0, c, false, false);
        c = __builtin_amdgcn_wmma_f32_16x16x32_bf16(false, a1, false, b1, (short)0, c, false, false);
        insertC(c, tile * 16 + (lane & 15));
    }

    const int half = lane >> 4, col = lane & 15;
    float tv[8]; int ti[8];
#pragma unroll
    for (int j = 0; j < 8; ++j) { tv[j] = -__builtin_inff(); ti[j] = 0; }

    // two-phase merge: waves 0-3 then waves 4-7 (keeps LDS at 64 KB)
#pragma unroll
    for (int phase = 0; phase < 2; ++phase) {
        if ((wave >> 2) == phase) {
            const int slot = wave & 3;
#pragma unroll
            for (int r = 0; r < 8; ++r) {
                const int row = half * 8 + r;  // C layout: VGPR r = row r / r+8
                const int base = ((row * 4 + slot) * 16 + col) * 8;
#pragma unroll
                for (int j = 0; j < 8; ++j) { cv[base + j] = bv[r][j]; ci[base + j] = bi[r][j]; }
            }
        }
        __syncthreads();
        if (threadIdx.x < 16) {
            const float* pv = cv + threadIdx.x * 512;
            const int*   pi = ci + threadIdx.x * 512;
            for (int e = 0; e < 512; ++e) {
                const float v = pv[e];
                if (v > tv[7]) {
                    tv[7] = v; ti[7] = pi[e];
#pragma unroll
                    for (int j = 7; j > 0; --j) {
                        if (tv[j] > tv[j - 1]) {
                            float t0 = tv[j]; tv[j] = tv[j - 1]; tv[j - 1] = t0;
                            int   t1 = ti[j]; ti[j] = ti[j - 1]; ti[j - 1] = t1;
                        }
                    }
                }
            }
        }
        __syncthreads();
    }
    if (threadIdx.x < 16) {
        int* outp = topIdx + (rowtile * 16 + threadIdx.x) * 8;
#pragma unroll
        for (int j = 0; j < 8; ++j) outp[j] = ti[j];   // descending, like top_k
    }
}

// ---------------------------------------------------------------------------
// Kernel 4: gather top-8 source rows -> LDS, out = match @ Wo^T + bo.
// 1 block = 16 rows; thread t owns output cols {t, t+256, t+512} x 16 rows.
// ---------------------------------------------------------------------------
__global__ __launch_bounds__(256) void gather_out(const float* __restrict__ sProj,
                                                  const int* __restrict__ topIdx,
                                                  const float* __restrict__ Wo,
                                                  const float* __restrict__ bo,
                                                  float* __restrict__ out) {
    __shared__ float4 match[16 * 128];         // [16 rows][512 feats] = 32 KB
    const int rowtile = blockIdx.x;
    const int t = threadIdx.x;
    if (t < 128) {
        const int i = t >> 3, j = t & 7;
        const int qi = topIdx[(rowtile * 16 + i) * 8 + j];
        const float4* srow = (const float4*)(sProj + (size_t)qi * 64);
#pragma unroll
        for (int e = 0; e < 16; ++e) match[i * 128 + j * 16 + e] = srow[e];
    }
    __syncthreads();

    const float4* Wo4 = (const float4*)Wo;
    float acc[3][16];
#pragma unroll
    for (int jj = 0; jj < 3; ++jj)
#pragma unroll
        for (int i = 0; i < 16; ++i) acc[jj][i] = 0.f;

    for (int k4 = 0; k4 < 128; ++k4) {
        const float4 w0 = Wo4[(size_t)(t)       * 128 + k4];
        const float4 w1 = Wo4[(size_t)(t + 256) * 128 + k4];
        const float4 w2 = Wo4[(size_t)(t + 512) * 128 + k4];
#pragma unroll
        for (int i = 0; i < 16; ++i) {
            const float4 m = match[i * 128 + k4];
            acc[0][i] = fmaf(w0.x, m.x, fmaf(w0.y, m.y, fmaf(w0.z, m.z, fmaf(w0.w, m.w, acc[0][i]))));
            acc[1][i] = fmaf(w1.x, m.x, fmaf(w1.y, m.y, fmaf(w1.z, m.z, fmaf(w1.w, m.w, acc[1][i]))));
            acc[2][i] = fmaf(w2.x, m.x, fmaf(w2.y, m.y, fmaf(w2.z, m.z, fmaf(w2.w, m.w, acc[2][i]))));
        }
    }
#pragma unroll
    for (int jj = 0; jj < 3; ++jj) {
        const int o = t + jj * 256;
        const float bias = bo[o];
#pragma unroll
        for (int i = 0; i < 16; ++i)
            out[(size_t)(rowtile * 16 + i) * 768 + o] = acc[jj][i] + bias;
    }
}

// ---------------------------------------------------------------------------
extern "C" void kernel_launch(void* const* d_in, const int* in_sizes, int n_in,
                              void* d_out, int out_size, void* d_ws, size_t ws_size,
                              hipStream_t stream) {
    const float* tgt = (const float*)d_in[0];  // [4,512,512]
    const float* src = (const float*)d_in[1];  // [50000,768]
    const float* Wt  = (const float*)d_in[2];  // [64,512]
    const float* bt  = (const float*)d_in[3];  // [64]
    const float* Ws  = (const float*)d_in[4];  // [64,768]
    const float* bs  = (const float*)d_in[5];  // [64]
    const float* Wo  = (const float*)d_in[6];  // [768,512]
    const float* bo  = (const float*)d_in[7];  // [768]
    float* out = (float*)d_out;                // [4,512,768]

    char* ws = (char*)d_ws;
    float*          sProj  = (float*)(ws + OFF_SPROJ);
    unsigned short* sNorm  = (unsigned short*)(ws + OFF_SNORM);
    unsigned short* tNorm  = (unsigned short*)(ws + OFF_TNORM);
    int*            topIdx = (int*)(ws + OFF_TOPIDX);

    proj_source<<<Q_,      64, 0, stream>>>(src, Ws, bs, sProj, sNorm);
    proj_target<<<R_,      64, 0, stream>>>(tgt, Wt, bt, tNorm);
    sim_topk   <<<R_ / 16, 256, 0, stream>>>(tNorm, sNorm, topIdx);
    gather_out <<<R_ / 16, 256, 0, stream>>>(sProj, topIdx, Wo, bo, out);
}
